// Gate_7988639170881
// MI455X (gfx1250) — compile-verified, matrix operand order checked by reading, not run on hardware
//
#include <hip/hip_runtime.h>

// ---------------------------------------------------------------------------
// DeepSeek-V3 style MoE gate for MI455X (gfx1250, wave32, WMMA).
//   scores  = sigmoid(x[16384,7168] @ W[256,7168]^T + bias)
//   routing = group-top2-sum -> top4 groups -> top8 experts -> renorm * 2.5
// GEMM via v_wmma_f32_16x16x32_bf16; W pre-converted to bf16 in d_ws.
// Memory-bound on streaming x once (470 MB -> ~20us @ 23.3 TB/s).
// k-loop unrolled x2 with ping-pong buffers: buffer index is a compile-time
// constant, so double-buffer "rotation" is pure register renaming (no
// v_mov churn), and WMMAs overlap the other buffer's loads.
// ---------------------------------------------------------------------------

#define T_TOK   16384
#define D_DIM   7168
#define E_EXP   256
#define NGRP    8
#define GSZ     32          // experts per group
#define TOPKG   4
#define TOPK    8
#define RSCALE  2.5f
#define M_BLK   32          // tokens per workgroup
#define NT_W    4           // 16-col N tiles per wave

typedef unsigned short u16;
typedef unsigned int   u32;
typedef __bf16 bf16;
typedef float v4f  __attribute__((ext_vector_type(4)));
typedef float v8f  __attribute__((ext_vector_type(8)));
typedef u16  v4u16 __attribute__((ext_vector_type(4)));
typedef u16  v8u16 __attribute__((ext_vector_type(8)));
typedef u16  v16u16 __attribute__((ext_vector_type(16)));
typedef bf16 v16bf __attribute__((ext_vector_type(16)));

// XOR-swizzled LDS index for a [32][256] f32 score tile (32 KB).
static __device__ __forceinline__ int sidx(int row, int e) {
  return row * E_EXP + (e ^ ((row & 31) << 1));
}

// --------------------------- W fp32 -> bf16 -------------------------------
__global__ void wcvt_kernel(const float* __restrict__ w,
                            u16* __restrict__ o, int n) {
  int i = (blockIdx.x * blockDim.x + threadIdx.x) * 4;
  if (i + 3 < n) {
    v4f f = *(const v4f*)(w + i);
    v4u16 b;
    b.x = __builtin_bit_cast(u16, (bf16)f.x);
    b.y = __builtin_bit_cast(u16, (bf16)f.y);
    b.z = __builtin_bit_cast(u16, (bf16)f.z);
    b.w = __builtin_bit_cast(u16, (bf16)f.w);
    *(v4u16*)(o + i) = b;
  }
}

// ------------------------ fused score + route -----------------------------
__global__ __launch_bounds__(256) void gate_kernel(
    const float* __restrict__ x,      // [T, D] f32
    const u16*   __restrict__ wbf,    // [E, D] bf16 bits
    const float* __restrict__ bias,   // [E]
    float*       __restrict__ out_w,  // [T, 8]
    int*         __restrict__ out_i)  // [T, 8]
{
  __shared__ float sc[M_BLK * E_EXP];          // 32 KB, swizzled

  const int tid   = threadIdx.x;
  const int wave  = tid >> 5;
  const int lane  = tid & 31;
  const int mtile = wave >> 2;                 // 0..1  (16-row M tile)
  const int nquad = wave & 3;                  // 0..3  (64-expert N quad)
  const int l15   = lane & 15;
  const int ksel  = (lane >> 4) << 3;          // K sub-segment: 0 or 8
  const int rowA  = (mtile << 4) + l15;
  const long tok  = (long)blockIdx.x * M_BLK + rowA;
  const float* xp = x + tok * (long)D_DIM + ksel;

  // Per-lane base pointers into the 4 expert rows this lane touches.
  const u16* wp[NT_W];
  #pragma unroll
  for (int nt = 0; nt < NT_W; ++nt) {
    int n = (nquad << 6) + (nt << 4) + l15;    // expert index = B column
    wp[nt] = wbf + (size_t)n * D_DIM + ksel;
  }

  v8f acc[NT_W] = {};                          // 4 x (16x16 f32) accumulators

  // Ping-pong buffers: [buf][...] with buf always a constant index.
  v4f   fA[2][4];
  v8u16 b0[2][NT_W], b1[2][NT_W];

  // Load stage: fill buffer `buf` with A floats + B fragments for k = kk.
  #define LOAD_STAGE(buf, kk)                                              \
    do {                                                                   \
      fA[buf][0] = *(const v4f*)(xp + (kk));                               \
      fA[buf][1] = *(const v4f*)(xp + (kk) + 4);                           \
      fA[buf][2] = *(const v4f*)(xp + (kk) + 16);                          \
      fA[buf][3] = *(const v4f*)(xp + (kk) + 20);                          \
      _Pragma("unroll")                                                    \
      for (int nt = 0; nt < NT_W; ++nt) {                                  \
        b0[buf][nt] = *(const v8u16*)(wp[nt] + (kk));                      \
        b1[buf][nt] = *(const v8u16*)(wp[nt] + (kk) + 16);                 \
      }                                                                    \
    } while (0)

  // WMMA stage: convert buffer `buf`'s A to bf16, run the 4 WMMAs.
  #define WMMA_STAGE(buf, afrag)                                           \
    do {                                                                   \
      _Pragma("unroll")                                                    \
      for (int i = 0; i < 4; ++i) {                                        \
        afrag[i]      = (bf16)fA[buf][0][i];                               \
        afrag[4 + i]  = (bf16)fA[buf][1][i];                               \
        afrag[8 + i]  = (bf16)fA[buf][2][i];                               \
        afrag[12 + i] = (bf16)fA[buf][3][i];                               \
      }                                                                    \
      _Pragma("unroll")                                                    \
      for (int nt = 0; nt < NT_W; ++nt) {                                  \
        v16u16 bu;                                                         \
        _Pragma("unroll")                                                  \
        for (int i = 0; i < 8; ++i) {                                      \
          bu[i] = b0[buf][nt][i]; bu[8 + i] = b1[buf][nt][i];              \
        }                                                                  \
        v16bf bmat = __builtin_bit_cast(v16bf, bu);                        \
        acc[nt] = __builtin_amdgcn_wmma_f32_16x16x32_bf16(                 \
            false, afrag, false, bmat, (short)0, acc[nt], false, false);   \
      }                                                                    \
    } while (0)

  // Prologue: buffer 0 holds k = 0.
  LOAD_STAGE(0, 0);

  // 7168 / 64 = 112 iterations; each handles two 32-wide k-steps.
  #pragma unroll 1
  for (int k0 = 0; k0 < D_DIM; k0 += 64) {
    // half 0: load buf1 @ k0+32 (always in range), consume buf0 @ k0
    v16bf a0;
    LOAD_STAGE(1, k0 + 32);
    WMMA_STAGE(0, a0);

    // half 1: load buf0 @ k0+64 (wraps to 0 on last iter; harmless reload),
    //         consume buf1 @ k0+32
    v16bf a1;
    const int k2 = (k0 + 64 < D_DIM) ? (k0 + 64) : 0;
    LOAD_STAGE(0, k2);
    WMMA_STAGE(1, a1);
  }

  #undef LOAD_STAGE
  #undef WMMA_STAGE

  // Epilogue: bias + sigmoid, scatter into swizzled LDS score tile.
  // C layout: VGPR r -> M = r (+8 for lanes 16..31), N = lane & 15.
  const int rowhi = (lane >> 4) << 3;
  #pragma unroll
  for (int nt = 0; nt < NT_W; ++nt) {
    int col = (nquad << 6) + (nt << 4) + l15;
    float bv = bias[col];
    #pragma unroll
    for (int r = 0; r < 8; ++r) {
      int row = (mtile << 4) + rowhi + r;
      float v = acc[nt][r] + bv;
      float s = 1.0f / (1.0f + __expf(-v));
      sc[sidx(row, col)] = s;
    }
  }
  __syncthreads();

  // ---------------- routing: one thread per token -------------------------
  if (tid < M_BLK) {
    const int t = tid;

    // group scores = sum of top-2 per group of 32
    float gs[NGRP];
    #pragma unroll 1
    for (int g = 0; g < NGRP; ++g) {
      float m1 = -1e30f, m2 = -1e30f;
      int base = g * GSZ;
      #pragma unroll 1
      for (int j = 0; j < GSZ; ++j) {
        float v = sc[sidx(t, base + j)];
        if (v > m1) { m2 = m1; m1 = v; }
        else if (v > m2) { m2 = v; }
      }
      gs[g] = m1 + m2;
    }

    // top-4 groups (strict > keeps lowest index on ties, like lax.top_k)
    bool keep[NGRP];
    #pragma unroll
    for (int g = 0; g < NGRP; ++g) keep[g] = false;
    #pragma unroll 1
    for (int it = 0; it < TOPKG; ++it) {
      float best = -1e30f; int bi = 0;
      #pragma unroll
      for (int g = 0; g < NGRP; ++g)
        if (!keep[g] && gs[g] > best) { best = gs[g]; bi = g; }
      keep[bi] = true;
    }

    // top-8 experts within kept groups; gathered weight = original score
    float wsel[TOPK]; int isel[TOPK];
    float sum = 0.0f;
    #pragma unroll 1
    for (int it = 0; it < TOPK; ++it) {
      float best = -1e30f; int bi = 0;
      #pragma unroll 1
      for (int g = 0; g < NGRP; ++g) {
        if (!keep[g]) continue;
        int base = g * GSZ;
        #pragma unroll 1
        for (int j = 0; j < GSZ; ++j) {
          float v = sc[sidx(t, base + j)];
          if (v > best) { best = v; bi = base + j; }
        }
      }
      wsel[it] = best;
      isel[it] = bi;
      sum += best;
      sc[sidx(t, bi)] = -1e30f;   // remove from further selection
    }

    const long gtok = (long)blockIdx.x * M_BLK + t;
    float inv = RSCALE / sum;     // sigmoid scores > 0, sum > 0
    #pragma unroll
    for (int it = 0; it < TOPK; ++it) {
      out_w[gtok * TOPK + it] = wsel[it] * inv;
      out_i[gtok * TOPK + it] = isel[it];
    }
  }
}

// ---------------------------------------------------------------------------
extern "C" void kernel_launch(void* const* d_in, const int* in_sizes, int n_in,
                              void* d_out, int out_size, void* d_ws, size_t ws_size,
                              hipStream_t stream) {
  (void)in_sizes; (void)n_in; (void)out_size; (void)ws_size;

  const float* x    = (const float*)d_in[0];   // [T, D]
  const float* w    = (const float*)d_in[1];   // [E, D]
  const float* bias = (const float*)d_in[2];   // [E]

  u16*   wbf   = (u16*)d_ws;                   // [E, D] bf16 (3.67 MB)
  float* out_w = (float*)d_out;                // [T, 8] f32
  int*   out_i = (int*)((float*)d_out + (size_t)T_TOK * TOPK);  // [T, 8] i32

  const int n = E_EXP * D_DIM;                 // 1,835,008 (divisible by 1024)
  wcvt_kernel<<<n / (256 * 4), 256, 0, stream>>>(w, wbf, n);

  gate_kernel<<<T_TOK / M_BLK, 256, 0, stream>>>(x, wbf, bias, out_w, out_i);
}
